// DeformConv2dBlock_69784628625710
// MI455X (gfx1250) — compile-verified, matrix operand order checked by reading, not run on hardware
//
#include <hip/hip_runtime.h>
#include <cstdint>
#include <cstddef>

// ---------------------------------------------------------------------------
// Deformable conv block for MI455X (gfx1250, wave32, WMMA).
// Pipeline: offset conv -> NCHW->NHWC transpose -> bilinear sample to f16
// im2col matrix A[M][K] -> WMMA f16 GEMM (M=36864, N=128, K=1152) with
// async-to-LDS double-buffered B staging and batched fragment loads
// -> BN reduction -> normalize+ReLU.
// ---------------------------------------------------------------------------

typedef _Float16 v4h  __attribute__((ext_vector_type(4)));
typedef _Float16 v8h  __attribute__((ext_vector_type(8)));
typedef _Float16 v16h __attribute__((ext_vector_type(16)));
typedef float    v4f  __attribute__((ext_vector_type(4)));
typedef float    v8f  __attribute__((ext_vector_type(8)));

#define N_  4
#define C_  128
#define O_  128
#define H_  96
#define W_  96
#define HW_ (H_*W_)        // 9216
#define M_  (N_*HW_)       // 36864
#define K_  (C_*9)         // 1152
#define BN_EPS 1e-5f

// Workspace layout (bytes, all 256B aligned). Total ~133.6 MB.
#define WS_OFF    ((size_t)0)                       // off: N*18*H*W f32 = 2.7 MB (10.6 MB slot)
#define WS_XNHWC  ((size_t)10616832)                // x NHWC f32      = 18.9 MB
#define WS_BPACK  ((size_t)29491200)                // B[o][K] f16     = 0.3 MB
#define WS_ABUF   ((size_t)29786112)                // A[M][K] f16     = 84.9 MB
#define WS_GBUF   ((size_t)114720768)               // GEMM out f32    = 18.9 MB
#define WS_SUMS   ((size_t)133595136)               // sum[128]+sumsq[128] f32

// ---------------------------------------------------------------------------
// 1) Offset conv: 3x3, 128 -> 18 channels, pad 1.  One thread per pixel,
//    all 18 output channels in registers; weights staged 32-ci at a time
//    through LDS (20.7 KB).
// ---------------------------------------------------------------------------
__global__ __launch_bounds__(256) void k_offset_conv(
    const float* __restrict__ x, const float* __restrict__ w_off,
    const float* __restrict__ b_off, float* __restrict__ off) {
  __shared__ float wlds[18 * 32 * 9];     // 5184 floats
  const int tid = threadIdx.x;
  const int pix = blockIdx.x * 256 + tid; // < 36864 exactly
  const int n = pix / HW_;
  const int hw = pix % HW_;
  const int h = hw / W_;
  const int w = hw % W_;

  float acc[18];
#pragma unroll
  for (int co = 0; co < 18; ++co) acc[co] = b_off[co];

  for (int cc = 0; cc < C_; cc += 32) {
    __syncthreads();
    for (int i = tid; i < 18 * 32 * 9; i += 256) {
      const int co = i / 288;
      const int r  = i % 288;
      const int c2 = r / 9;
      const int q  = r % 9;
      wlds[i] = w_off[(size_t)co * K_ + (size_t)(cc + c2) * 9 + q];
    }
    __syncthreads();

    for (int c2 = 0; c2 < 32; ++c2) {
      const int ci = cc + c2;
      const float* xp = x + (((size_t)n * C_ + ci) * H_) * W_;
      float tap[9];
#pragma unroll
      for (int ky = 0; ky < 3; ++ky) {
        const int y = h + ky - 1;
#pragma unroll
        for (int kx = 0; kx < 3; ++kx) {
          const int xx = w + kx - 1;
          const bool ok = (y >= 0) & (y < H_) & (xx >= 0) & (xx < W_);
          tap[ky * 3 + kx] = ok ? xp[y * W_ + xx] : 0.f;
        }
      }
#pragma unroll
      for (int q = 0; q < 9; ++q) {
        const float t = tap[q];
#pragma unroll
        for (int co = 0; co < 18; ++co)
          acc[co] = fmaf(t, wlds[co * 288 + c2 * 9 + q], acc[co]);
      }
    }
  }
#pragma unroll
  for (int co = 0; co < 18; ++co)
    off[(((size_t)n * 18 + co) * H_ + h) * W_ + w] = acc[co];
}

// ---------------------------------------------------------------------------
// 2) NCHW -> NHWC transpose via 32x32 LDS tiles (coalesced both sides).
// ---------------------------------------------------------------------------
__global__ __launch_bounds__(256) void k_transpose(
    const float* __restrict__ x, float* __restrict__ xn) {
  __shared__ float tile[32][33];
  const int bid = blockIdx.x;
  const int wt = bid % 3;
  const int ct = (bid / 3) % 4;
  const int nh = bid / 12;
  const int n = nh / H_;
  const int h = nh % H_;
  const int tid = threadIdx.x;

#pragma unroll
  for (int p = 0; p < 4; ++p) {
    const int li = tid + p * 256;
    const int ci = li >> 5;
    const int wi = li & 31;
    tile[ci][wi] =
        x[(((size_t)n * C_ + ct * 32 + ci) * H_ + h) * W_ + wt * 32 + wi];
  }
  __syncthreads();
#pragma unroll
  for (int p = 0; p < 4; ++p) {
    const int li = tid + p * 256;
    const int wi = li >> 5;
    const int ci = li & 31;
    xn[(((size_t)n * H_ + h) * W_ + wt * 32 + wi) * C_ + ct * 32 + ci] =
        tile[ci][wi];
  }
}

// ---------------------------------------------------------------------------
// 3) Pack w_def (O,C,3,3) f32 -> B[o][k*128+ci] f16 (K-contiguous rows).
// ---------------------------------------------------------------------------
__global__ __launch_bounds__(256) void k_packB(
    const float* __restrict__ w_def, _Float16* __restrict__ B) {
  const int idx = blockIdx.x * 256 + threadIdx.x;  // < 128*1152 exactly
  const int o = idx / K_;
  const int r = idx % K_;
  const int k = r / C_;
  const int ci = r % C_;
  B[idx] = (_Float16)w_def[(size_t)o * K_ + (size_t)ci * 9 + k];
}

// ---------------------------------------------------------------------------
// 4) Bilinear sampling.  One wave per (pixel m, kernel tap k); each lane
//    handles 4 channels (float4 coalesced corner loads from NHWC, packed
//    v4h coalesced store into A[m][k*128+ci]).  Control flow is wave-uniform.
// ---------------------------------------------------------------------------
__device__ inline void fma4(v4f& r, const float* p, float wgt) {
  const v4f v = *(const v4f*)p;
  r[0] = fmaf(v[0], wgt, r[0]);
  r[1] = fmaf(v[1], wgt, r[1]);
  r[2] = fmaf(v[2], wgt, r[2]);
  r[3] = fmaf(v[3], wgt, r[3]);
}

__global__ __launch_bounds__(256) void k_sample(
    const float* __restrict__ off, const float* __restrict__ xn,
    _Float16* __restrict__ A) {
  const int wave = threadIdx.x >> 5;
  const int lane = threadIdx.x & 31;
  const int wid = blockIdx.x * 8 + wave;     // < 36864*9 exactly
  const int m = wid / 9;
  const int k = wid % 9;
  const int n = m / HW_;
  const int hw = m % HW_;
  const int h = hw / W_;
  const int w = hw % W_;

  const float dy = off[(((size_t)n * 18 + 2 * k) * H_ + h) * W_ + w];
  const float dx = off[(((size_t)n * 18 + 2 * k + 1) * H_ + h) * W_ + w];
  const int ky = k / 3, kx = k % 3;
  const float py = dy + (float)(h + ky - 1);
  const float px = dx + (float)(w + kx - 1);
  const float y0f = floorf(py), x0f = floorf(px);
  const float wy = py - y0f, wx = px - x0f;
  const int y0 = (int)y0f, x0 = (int)x0f;
  const float w00 = (1.f - wy) * (1.f - wx);
  const float w01 = (1.f - wy) * wx;
  const float w10 = wy * (1.f - wx);
  const float w11 = wy * wx;

  const float* xb = xn + (size_t)n * HW_ * C_ + (size_t)lane * 4;
  v4f r = {0.f, 0.f, 0.f, 0.f};
  if (y0 >= 0 && y0 < H_ && x0 >= 0 && x0 < W_)
    fma4(r, xb + ((size_t)y0 * W_ + x0) * C_, w00);
  if (y0 >= 0 && y0 < H_ && x0 + 1 >= 0 && x0 + 1 < W_)
    fma4(r, xb + ((size_t)y0 * W_ + x0 + 1) * C_, w01);
  if (y0 + 1 >= 0 && y0 + 1 < H_ && x0 >= 0 && x0 < W_)
    fma4(r, xb + ((size_t)(y0 + 1) * W_ + x0) * C_, w10);
  if (y0 + 1 >= 0 && y0 + 1 < H_ && x0 + 1 >= 0 && x0 + 1 < W_)
    fma4(r, xb + ((size_t)(y0 + 1) * W_ + x0 + 1) * C_, w11);

  v4h o2;
  o2[0] = (_Float16)r[0];
  o2[1] = (_Float16)r[1];
  o2[2] = (_Float16)r[2];
  o2[3] = (_Float16)r[3];
  *(v4h*)(A + (size_t)m * K_ + k * C_ + lane * 4) = o2;
}

// ---------------------------------------------------------------------------
// 5) WMMA GEMM:  G[M][128] = A[M][K] * B[128][K]^T, f16 in / f32 accumulate.
//    288 blocks x 8 waves; wave owns rows [blk*128 + wave*16, +16) x all 128
//    output channels (8 accumulator tiles).
//    B: async-copied into double-buffered LDS (K-chunk 64, padded row stride
//       72 halves for conflict-free fragment reads).  Per k-step, ALL 8
//    fragments are loaded into registers first (16 clustered ds_load_b128),
//    then 8 back-to-back WMMAs — a scheduling fence keeps the loads above
//    the WMMA run so waits become partial (in-order DS returns) and one LDS
//    round trip is amortized over 8 WMMAs.
// ---------------------------------------------------------------------------
#define BROW 72                     // padded LDS row stride in halves
#define KCH  64                     // K per chunk
#define NCH  (K_ / KCH)             // 18 chunks

__device__ __forceinline__ void async_load_b128(uint32_t lds_off,
                                                const void* gptr) {
  asm volatile("global_load_async_to_lds_b128 %0, %1, off"
               :
               : "v"(lds_off), "v"((unsigned long long)(uintptr_t)gptr)
               : "memory");
}
__device__ __forceinline__ void wait_async0() {
  asm volatile("s_wait_asynccnt 0x0" ::: "memory");
}
// Scheduling fence: memory ops may not cross downward past this point.
__device__ __forceinline__ void sched_fence() {
  asm volatile("" ::: "memory");
}

// Thread tid stages 64 bytes of the B chunk [kk, kk+64) into LDS buffer.
__device__ __forceinline__ void stage_chunk(const _Float16* __restrict__ B,
                                            uint32_t ldsbase, int kk,
                                            int tid) {
  const int o  = tid >> 1;           // row 0..127
  const int hb = (tid & 1) * 32;     // half-offset within row: 0 or 32
  const _Float16* src = B + (size_t)o * K_ + kk + hb;
  const uint32_t dst = ldsbase + (uint32_t)(o * (BROW * 2) + hb * 2);
#pragma unroll
  for (int j = 0; j < 4; ++j)
    async_load_b128(dst + j * 16, src + j * 8);
}

// Fragment from LDS: column `col`, k-step s within chunk, lane half `khalf`.
__device__ __forceinline__ v16h lds_frag(const _Float16* base, int col, int s,
                                         int khalf) {
  const _Float16* p = base + col * BROW + s * 32 + khalf;
  const v8h lo = *(const v8h*)(p);
  const v8h hi = *(const v8h*)(p + 16);
  return __builtin_shufflevector(lo, hi, 0, 1, 2, 3, 4, 5, 6, 7, 8, 9, 10, 11,
                                 12, 13, 14, 15);
}

// Fragment from global A at K offset kk.
__device__ __forceinline__ v16h gl_frag(const _Float16* arow, int kk) {
  const v8h lo = *(const v8h*)(arow + kk);
  const v8h hi = *(const v8h*)(arow + kk + 16);
  return __builtin_shufflevector(lo, hi, 0, 1, 2, 3, 4, 5, 6, 7, 8, 9, 10, 11,
                                 12, 13, 14, 15);
}

__global__ __launch_bounds__(256) void k_gemm_wmma(
    const _Float16* __restrict__ A, const _Float16* __restrict__ B,
    float* __restrict__ G) {
  __shared__ _Float16 Bs[2][128 * BROW];   // 2 x 18 KB
  const int tid = threadIdx.x;
  const int wave = tid >> 5;
  const int lane = tid & 31;
  const int rlo = lane & 15;
  const int khalf = (lane >> 4) * 8;
  const int row0 = blockIdx.x * 128 + wave * 16;
  const uint32_t lds0 = (uint32_t)(uintptr_t)(&Bs[0][0]);
  const uint32_t lds1 = (uint32_t)(uintptr_t)(&Bs[1][0]);

  v8f acc[8];
#pragma unroll
  for (int t = 0; t < 8; ++t)
    acc[t] = (v8f){0.f, 0.f, 0.f, 0.f, 0.f, 0.f, 0.f, 0.f};

  // A fragment row base; per ISA 7.12.2: lane L holds row (L&15),
  // K-halves [kb..kb+7] and [kb+16..kb+23] with kb = 8*(L>>4).
  const _Float16* arow = A + (size_t)(row0 + rlo) * K_ + khalf;
  v16h a_cur = gl_frag(arow, 0);

  stage_chunk(B, lds0, 0, tid);   // prologue: chunk 0 -> buf0
  wait_async0();
  __syncthreads();

  for (int c = 0; c < NCH; ++c) {
    const int cur = c & 1;
    const _Float16* bsc = &Bs[cur][0];
    if (c + 1 < NCH)                       // overlap DMA with 16 WMMAs
      stage_chunk(B, cur ? lds0 : lds1, (c + 1) * KCH, tid);

    // Prefetch both upcoming A fragments (issued before the WMMA runs; the
    // per-step fences below keep these loads from sinking to their uses).
    v16h a_mid = gl_frag(arow, c * KCH + 32);          // always in range
    v16h a_nx0 = a_cur;
    if (c + 1 < NCH) a_nx0 = gl_frag(arow, (c + 1) * KCH);

#pragma unroll
    for (int s = 0; s < 2; ++s) {
      // Load ALL 8 B fragments of this k-step (16 ds_load_b128, clustered).
      v16h bf[8];
#pragma unroll
      for (int t = 0; t < 8; ++t)
        bf[t] = lds_frag(bsc, t * 16 + rlo, s, khalf);
      sched_fence();   // pin the loads above the WMMA run
      const v16h av = (s == 0) ? a_cur : a_mid;
#pragma unroll
      for (int t = 0; t < 8; ++t)
        acc[t] = __builtin_amdgcn_wmma_f32_16x16x32_f16(
            false, av, false, bf[t], (short)0, acc[t], false, false);
    }

    a_cur = a_nx0;
    if (c + 1 < NCH) {
      wait_async0();       // staged chunk fully in LDS
      __syncthreads();     // ...and all waves done reading 'cur'
    }
  }

  // D layout: VGPR v -> row (v + 8*(lane>>4)), col (lane&15).
  const int radd = (lane >> 4) * 8;
#pragma unroll
  for (int t = 0; t < 8; ++t) {
#pragma unroll
    for (int v = 0; v < 8; ++v) {
      const int m = row0 + v + radd;
      const int o = t * 16 + rlo;
      G[(size_t)m * O_ + o] = acc[t][v];
    }
  }
}

// ---------------------------------------------------------------------------
// 6) BN statistics: zero, then per-block LDS partial sums + f32 atomics.
//    (b_def shifts mean identically and cancels in BN; omitted analytically.)
// ---------------------------------------------------------------------------
__global__ __launch_bounds__(256) void k_zero_sums(float* __restrict__ s) {
  s[threadIdx.x] = 0.f;  // sum[0..127], sumsq[128..255]
}

__global__ __launch_bounds__(256) void k_reduce(
    const float* __restrict__ G, float* __restrict__ sums) {
  __shared__ float sS[256];
  __shared__ float sQ[256];
  const int tid = threadIdx.x;
  const int c = tid & 127;
  const int hf = tid >> 7;
  const int mbase = blockIdx.x * 256;
  float s = 0.f, q = 0.f;
  for (int r = hf; r < 256; r += 2) {
    const float v = G[(size_t)(mbase + r) * O_ + c];
    s += v;
    q = fmaf(v, v, q);
  }
  sS[tid] = s;
  sQ[tid] = q;
  __syncthreads();
  if (tid < 128) {
    atomicAdd(&sums[c], sS[tid] + sS[tid + 128]);
    atomicAdd(&sums[128 + c], sQ[tid] + sQ[tid + 128]);
  }
}

// ---------------------------------------------------------------------------
// 7) Normalize + affine + ReLU, write NCHW output.
// ---------------------------------------------------------------------------
__global__ __launch_bounds__(256) void k_finalize(
    const float* __restrict__ G, const float* __restrict__ sums,
    const float* __restrict__ gamma, const float* __restrict__ beta,
    float* __restrict__ out) {
  const int idx = blockIdx.x * 256 + threadIdx.x;  // < N*O*H*W exactly
  const int n = idx / (O_ * HW_);
  const int rem = idx % (O_ * HW_);
  const int o = rem / HW_;
  const int hw = rem % HW_;
  const int m = n * HW_ + hw;
  const float inv = 1.f / (float)M_;
  const float mean = sums[o] * inv;
  const float var = fmaf(-mean, mean, sums[128 + o] * inv);
  const float v = G[(size_t)m * O_ + o];
  float r = gamma[o] * (v - mean) * rsqrtf(var + BN_EPS) + beta[o];
  out[idx] = fmaxf(r, 0.f);
}

// ---------------------------------------------------------------------------
extern "C" void kernel_launch(void* const* d_in, const int* in_sizes, int n_in,
                              void* d_out, int out_size, void* d_ws,
                              size_t ws_size, hipStream_t stream) {
  const float* x     = (const float*)d_in[0];
  const float* w_off = (const float*)d_in[1];
  const float* b_off = (const float*)d_in[2];
  const float* w_def = (const float*)d_in[3];
  // d_in[4] = b_def: cancels exactly inside BatchNorm (shifts mean only).
  const float* gamma = (const float*)d_in[5];
  const float* beta  = (const float*)d_in[6];
  float* out = (float*)d_out;

  char* ws = (char*)d_ws;
  float*    off = (float*)(ws + WS_OFF);
  float*    xn  = (float*)(ws + WS_XNHWC);
  _Float16* B   = (_Float16*)(ws + WS_BPACK);
  _Float16* A   = (_Float16*)(ws + WS_ABUF);
  float*    G   = (float*)(ws + WS_GBUF);
  float*    sums = (float*)(ws + WS_SUMS);

  k_offset_conv<<<dim3(M_ / 256), dim3(256), 0, stream>>>(x, w_off, b_off, off);
  k_transpose<<<dim3(3 * 4 * N_ * H_), dim3(256), 0, stream>>>(x, xn);
  k_packB<<<dim3((O_ * K_) / 256), dim3(256), 0, stream>>>(w_def, B);
  k_sample<<<dim3((M_ * 9) / 8), dim3(256), 0, stream>>>(off, xn, A);
  k_gemm_wmma<<<dim3(M_ / 128), dim3(256), 0, stream>>>(A, B, G);
  k_zero_sums<<<dim3(1), dim3(256), 0, stream>>>(sums);
  k_reduce<<<dim3(M_ / 256), dim3(256), 0, stream>>>(G, sums);
  k_finalize<<<dim3((N_ * O_ * HW_) / 256), dim3(256), 0, stream>>>(
      G, sums, gamma, beta, out);
}
